// MultiHeadAttention_28114855919961
// MI455X (gfx1250) — matrix operand order repaired and checked by reading, not compiled, not run on hardware
//
#include <hip/hip_runtime.h>
#include <hip/hip_bf16.h>
#include <math.h>

// Problem constants (fixed by the reference).
constexpr int B_  = 4;
constexpr int T_  = 1024;
constexpr int C_  = 2048;
constexpr int H_  = 16;
constexpr int HD_ = 128;

typedef _Float16 v16h __attribute__((ext_vector_type(16)));
typedef _Float16 h8   __attribute__((ext_vector_type(8)));
typedef _Float16 h4   __attribute__((ext_vector_type(4)));
typedef float    v8f  __attribute__((ext_vector_type(8)));

__device__ __forceinline__ v8f zero_v8f() {
  v8f z;
#pragma unroll
  for (int i = 0; i < 8; ++i) z[i] = 0.0f;
  return z;
}

__device__ __forceinline__ v8f wmma_f16(v16h a, v16h b, v8f c) {
  return __builtin_amdgcn_wmma_f32_16x16x32_f16(false, a, false, b, (short)0, c,
                                                false, false);
}

// Async global->LDS copy (CDNA5, ASYNCcnt-tracked). lds_off is the 32-bit LDS
// byte offset (low 32 bits of the shared pointer), global addr a VGPR pair.
__device__ __forceinline__ void async_load_b128(unsigned lds_off,
                                                const _Float16* gptr) {
  asm volatile("global_load_async_to_lds_b128 %0, %1, off"
               :: "v"(lds_off), "v"((unsigned long long)(size_t)gptr)
               : "memory");
}
__device__ __forceinline__ void wait_async0() {
  asm volatile("s_wait_asynccnt 0" ::: "memory");
}

// A fragment: 16x32 f16 tile (row-major, leading dim `ld` halves).
// Lane L holds row r = L&15; K chunks [(L>>4)*8 .. +7] and [16+(L>>4)*8 .. +7].
__device__ __forceinline__ v16h load_a_frag(const _Float16* base, int ld, int lane) {
  const int r    = lane & 15;
  const int koff = (lane >> 4) * 8;
  const h8 lo = *reinterpret_cast<const h8*>(base + r * ld + koff);
  const h8 hi = *reinterpret_cast<const h8*>(base + r * ld + 16 + koff);
  v16h a;
#pragma unroll
  for (int i = 0; i < 8; ++i) { a[i] = lo[i]; a[8 + i] = hi[i]; }
  return a;
}

// B fragment: 32(K)x16(N) where memory holds B^T row-major (base[n*ld + k]).
// Lane L holds column n = L&15; K chunks [(L>>4)*8 .. +7] and [16+(L>>4)*8 .. +7].
__device__ __forceinline__ v16h load_b_frag(const _Float16* base, int ld, int lane) {
  const int n  = lane & 15;
  const int kb = (lane >> 4) * 8;
  const h8 lo = *reinterpret_cast<const h8*>(base + n * ld + kb);
  const h8 hi = *reinterpret_cast<const h8*>(base + n * ld + kb + 16);
  v16h b;
#pragma unroll
  for (int i = 0; i < 8; ++i) { b[i] = lo[i]; b[8 + i] = hi[i]; }
  return b;
}

// ---------------------------------------------------------------------------
// One-pass f32 -> f16 conversion (x, w_attn, w_proj), so the GEMMs consume
// f16 directly and can stage tiles with async global->LDS copies.
// ---------------------------------------------------------------------------
__global__ __launch_bounds__(256) void cvt_kernel(const float* __restrict__ s,
                                                  _Float16* __restrict__ d,
                                                  int n4) {
  const int i = blockIdx.x * 256 + threadIdx.x;
  if (i >= n4) return;
  const float4 v = reinterpret_cast<const float4*>(s)[i];
  h4 o;
  o[0] = (_Float16)v.x;
  o[1] = (_Float16)v.y;
  o[2] = (_Float16)v.z;
  o[3] = (_Float16)v.w;
  reinterpret_cast<h4*>(d)[i] = o;
}

// ---------------------------------------------------------------------------
// GEMM tiling: 128(M) x 128(N) block, 4 waves as 2x2, 64x64 per wave,
// K-step 32, double-buffered LDS fed by async global->LDS copies.
// ---------------------------------------------------------------------------
constexpr int BM  = 128;
constexpr int BN  = 128;
constexpr int BK  = 32;
constexpr int LDT = BK + 8;  // padded LDS stride (halves)

// GEMM1: qkv[f16, [3][B][H][T][HD]] = x[f16, M=B*T x K=C] * w_attn[f16, N=3C x K]^T
__global__ __launch_bounds__(128) void qkv_gemm_kernel(
    const _Float16* __restrict__ A, const _Float16* __restrict__ Bw,
    _Float16* __restrict__ qkv) {
  __shared__ _Float16 sA[2][BM * LDT];
  __shared__ _Float16 sB[2][BN * LDT];
  const int tid  = threadIdx.x;
  const int lane = tid & 31;
  const int w    = tid >> 5;
  const int wm   = w >> 1, wn = w & 1;
  const int row0 = blockIdx.y * BM;
  const int col0 = blockIdx.x * BN;
  const int K    = C_;

  // Stage one 128x32 A tile + 128x32 B tile via async copies (8 x b128/lane).
  auto stage = [&](int buf, int k0) {
#pragma unroll
    for (int it = 0; it < 4; ++it) {
      const int q = it * 128 + tid;
      const int r = q >> 2;
      const int c = (q & 3) * 8;
      async_load_b128((unsigned)(size_t)&sA[buf][r * LDT + c],
                      A + (size_t)(row0 + r) * K + k0 + c);
    }
#pragma unroll
    for (int it = 0; it < 4; ++it) {
      const int q = it * 128 + tid;
      const int r = q >> 2;
      const int c = (q & 3) * 8;
      async_load_b128((unsigned)(size_t)&sB[buf][r * LDT + c],
                      Bw + (size_t)(col0 + r) * K + k0 + c);
    }
  };

  v8f acc[4][4];
#pragma unroll
  for (int i = 0; i < 4; ++i)
#pragma unroll
    for (int j = 0; j < 4; ++j) acc[i][j] = zero_v8f();

  stage(0, 0);
  wait_async0();
  __syncthreads();

  constexpr int NT = C_ / BK;  // 64
  for (int kt = 0; kt < NT; ++kt) {
    const int cur = kt & 1;
    if (kt + 1 < NT) stage(1 - cur, (kt + 1) * BK);

    v16h af[4], bf[4];
#pragma unroll
    for (int i = 0; i < 4; ++i)
      af[i] = load_a_frag(sA[cur] + (wm * 64 + i * 16) * LDT, LDT, lane);
#pragma unroll
    for (int j = 0; j < 4; ++j)
      bf[j] = load_b_frag(sB[cur] + (wn * 64 + j * 16) * LDT, LDT, lane);
#pragma unroll
    for (int i = 0; i < 4; ++i)
#pragma unroll
      for (int j = 0; j < 4; ++j) acc[i][j] = wmma_f16(af[i], bf[j], acc[i][j]);

    if (kt + 1 < NT) {
      wait_async0();
      __syncthreads();
    }
  }

  // Scatter into qkv[three][b][h][t][d] (f16).
#pragma unroll
  for (int i = 0; i < 4; ++i)
#pragma unroll
    for (int j = 0; j < 4; ++j) {
      const int gn    = col0 + wn * 64 + j * 16 + (lane & 15);
      const int three = gn >> 11;         // / C_
      const int rem   = gn & (C_ - 1);
      const int h     = rem >> 7;         // / HD_
      const int d     = rem & (HD_ - 1);
#pragma unroll
      for (int e = 0; e < 8; ++e) {
        const int gm = row0 + wm * 64 + i * 16 + e + ((lane >> 4) << 3);
        const int b  = gm >> 10;          // / T_
        const int t  = gm & (T_ - 1);
        const size_t oidx =
            ((((size_t)three * B_ + b) * H_ + h) * T_ + t) * HD_ + d;
        qkv[oidx] = (_Float16)acc[i][j][e];
      }
    }
}

// GEMM3: out[f32, M=B*T x N=C] = y[f16, M x K=C] * w_proj[f16, N x K]^T
__global__ __launch_bounds__(128) void proj_gemm_kernel(
    const _Float16* __restrict__ A, const _Float16* __restrict__ Bw,
    float* __restrict__ out) {
  __shared__ _Float16 sA[2][BM * LDT];
  __shared__ _Float16 sB[2][BN * LDT];
  const int tid  = threadIdx.x;
  const int lane = tid & 31;
  const int w    = tid >> 5;
  const int wm   = w >> 1, wn = w & 1;
  const int row0 = blockIdx.y * BM;
  const int col0 = blockIdx.x * BN;
  const int K    = C_;

  auto stage = [&](int buf, int k0) {
#pragma unroll
    for (int it = 0; it < 4; ++it) {
      const int q = it * 128 + tid;
      const int r = q >> 2;
      const int c = (q & 3) * 8;
      async_load_b128((unsigned)(size_t)&sA[buf][r * LDT + c],
                      A + (size_t)(row0 + r) * K + k0 + c);
    }
#pragma unroll
    for (int it = 0; it < 4; ++it) {
      const int q = it * 128 + tid;
      const int r = q >> 2;
      const int c = (q & 3) * 8;
      async_load_b128((unsigned)(size_t)&sB[buf][r * LDT + c],
                      Bw + (size_t)(col0 + r) * K + k0 + c);
    }
  };

  v8f acc[4][4];
#pragma unroll
  for (int i = 0; i < 4; ++i)
#pragma unroll
    for (int j = 0; j < 4; ++j) acc[i][j] = zero_v8f();

  stage(0, 0);
  wait_async0();
  __syncthreads();

  constexpr int NT = C_ / BK;  // 64
  for (int kt = 0; kt < NT; ++kt) {
    const int cur = kt & 1;
    if (kt + 1 < NT) stage(1 - cur, (kt + 1) * BK);

    v16h af[4], bf[4];
#pragma unroll
    for (int i = 0; i < 4; ++i)
      af[i] = load_a_frag(sA[cur] + (wm * 64 + i * 16) * LDT, LDT, lane);
#pragma unroll
    for (int j = 0; j < 4; ++j)
      bf[j] = load_b_frag(sB[cur] + (wn * 64 + j * 16) * LDT, LDT, lane);
#pragma unroll
    for (int i = 0; i < 4; ++i)
#pragma unroll
      for (int j = 0; j < 4; ++j) acc[i][j] = wmma_f16(af[i], bf[j], acc[i][j]);

    if (kt + 1 < NT) {
      wait_async0();
      __syncthreads();
    }
  }

#pragma unroll
  for (int i = 0; i < 4; ++i)
#pragma unroll
    for (int j = 0; j < 4; ++j) {
      const int gn = col0 + wn * 64 + j * 16 + (lane & 15);
#pragma unroll
      for (int e = 0; e < 8; ++e) {
        const int gm = row0 + wm * 64 + i * 16 + e + ((lane >> 4) << 3);
        out[(size_t)gm * C_ + gn] = acc[i][j][e];
      }
    }
}

// ---------------------------------------------------------------------------
// Flash-style attention: one block per (b, h, 64-row q block); 4 waves,
// 16 q-rows per wave; KV blocks of 64 keys; online softmax in registers.
// K block staged with CDNA5 async global->LDS copies (ASYNCcnt).
// ---------------------------------------------------------------------------
constexpr int QB   = 64;        // q rows per block
constexpr int KB   = 64;        // keys per kv block
constexpr int LDK  = HD_ + 8;   // 136: K block stride (breaks 256B bank period)
constexpr int LDVT = KB + 8;    // 72: transposed-V stride
constexpr int LDP  = KB + 8;    // 72: per-wave P-tile stride

__global__ __launch_bounds__(128) void attention_kernel(
    const _Float16* __restrict__ qkv, const int* __restrict__ amask,
    _Float16* __restrict__ Y) {
  __shared__ _Float16 sK[KB * LDK];         // keys, [kv][d]
  __shared__ _Float16 sVT[HD_ * LDVT];      // values transposed, [d][kv]
  __shared__ _Float16 sP[4 * 16 * LDP];     // per-wave P tiles

  const int tid  = threadIdx.x;
  const int lane = tid & 31;
  const int w    = tid >> 5;
  const int blk  = blockIdx.x;
  const int qblk = blk & 15;            // T_/QB = 16
  const int h    = (blk >> 4) & 15;
  const int b    = blk >> 8;
  const int qbase = qblk * QB + w * 16;

  const _Float16* Q  = qkv + (((size_t)0 * B_ + b) * H_ + h) * (size_t)T_ * HD_;
  const _Float16* Kp = qkv + (((size_t)1 * B_ + b) * H_ + h) * (size_t)T_ * HD_;
  const _Float16* Vp = qkv + (((size_t)2 * B_ + b) * H_ + h) * (size_t)T_ * HD_;

  // Resident Q fragments: 16 rows x 128 K, 4 fragments of 16x32.
  v16h qf[4];
#pragma unroll
  for (int kc = 0; kc < 4; ++kc)
    qf[kc] = load_a_frag(Q + (size_t)qbase * HD_ + kc * 32, HD_, lane);

  v8f O[8];
#pragma unroll
  for (int nt = 0; nt < 8; ++nt) O[nt] = zero_v8f();
  float m[8], l[8];
#pragma unroll
  for (int e = 0; e < 8; ++e) { m[e] = -3.0e38f; l[e] = 0.0f; }

  const float scale = 0.08838834764831845f;  // 1/sqrt(128)
  _Float16* sPw = sP + w * 16 * LDP;

  for (int kv0 = 0; kv0 < T_; kv0 += KB) {
    // K block: async global->LDS (proceeds while we transpose V below).
#pragma unroll
    for (int it = 0; it < 8; ++it) {
      const int q = it * 128 + tid;
      const int r = q >> 4;
      const int c = (q & 15) * 8;
      async_load_b128((unsigned)(size_t)&sK[r * LDK + c],
                      Kp + (size_t)(kv0 + r) * HD_ + c);
    }
    // V block: load and transpose into sVT[d][kv].
#pragma unroll
    for (int it = 0; it < 8; ++it) {
      const int q = it * 128 + tid;
      const int r = q >> 4;
      const int c = (q & 15) * 8;
      const h8 vv = *reinterpret_cast<const h8*>(
          Vp + (size_t)(kv0 + r) * HD_ + c);
#pragma unroll
      for (int i = 0; i < 8; ++i) sVT[(c + i) * LDVT + r] = vv[i];
    }
    wait_async0();
    __syncthreads();

    // S = Q * K^T for this 16x64 tile.
    v8f S[4];
#pragma unroll
    for (int nt = 0; nt < 4; ++nt) S[nt] = zero_v8f();
#pragma unroll
    for (int nt = 0; nt < 4; ++nt)
#pragma unroll
      for (int kc = 0; kc < 4; ++kc) {
        const v16h bf = load_b_frag(sK + (nt * 16) * LDK + kc * 32, LDK, lane);
        S[nt] = wmma_f16(qf[kc], bf, S[nt]);
      }

    // Scale + padding-mask bias.
#pragma unroll
    for (int nt = 0; nt < 4; ++nt) {
      const int col    = kv0 + nt * 16 + (lane & 15);
      const float bias = (amask[b * T_ + col] == 0) ? -3.0e38f : 0.0f;
#pragma unroll
      for (int e = 0; e < 8; ++e) S[nt][e] = S[nt][e] * scale + bias;
    }

    // Online softmax: row-wise max/sum via 16-lane shuffle reductions.
#pragma unroll
    for (int e = 0; e < 8; ++e) {
      float v = fmaxf(fmaxf(S[0][e], S[1][e]), fmaxf(S[2][e], S[3][e]));
#pragma unroll
      for (int off = 8; off >= 1; off >>= 1) v = fmaxf(v, __shfl_xor(v, off, 32));
      const float mnew = fmaxf(m[e], v);

      const int lrow = e + ((lane >> 4) << 3);
      float rsum = 0.0f;
#pragma unroll
      for (int nt = 0; nt < 4; ++nt) {
        const float p = __expf(S[nt][e] - mnew);
        rsum += p;
        sPw[lrow * LDP + nt * 16 + (lane & 15)] = (_Float16)p;
      }
#pragma unroll
      for (int off = 8; off >= 1; off >>= 1) rsum += __shfl_xor(rsum, off, 32);

      const float alpha = __expf(m[e] - mnew);
      l[e] = l[e] * alpha + rsum;
      m[e] = mnew;
#pragma unroll
      for (int nt = 0; nt < 8; ++nt) O[nt][e] *= alpha;
    }
    __threadfence_block();  // order per-wave P stores before P fragment loads

    // O += P * V  (K-dim = 64 keys, split in two 32-wide fragments).
    v16h pf[2];
    pf[0] = load_a_frag(sPw, LDP, lane);
    pf[1] = load_a_frag(sPw + 32, LDP, lane);
#pragma unroll
    for (int nt = 0; nt < 8; ++nt)
#pragma unroll
      for (int kb = 0; kb < 2; ++kb) {
        const v16h vf =
            load_b_frag(sVT + (nt * 16) * LDVT + kb * 32, LDVT, lane);
        O[nt] = wmma_f16(pf[kb], vf, O[nt]);
      }
    __syncthreads();
  }

  // Normalize and write y[b][t][h*HD + d] as f16 for the final projection.
#pragma unroll
  for (int nt = 0; nt < 8; ++nt) {
    const int d = nt * 16 + (lane & 15);
#pragma unroll
    for (int e = 0; e < 8; ++e) {
      const int t = qbase + e + ((lane >> 4) << 3);
      const float val = O[nt][e] / l[e];
      Y[((size_t)b * T_ + t) * C_ + h * HD_ + d] = (_Float16)val;
    }
  }
}

extern "C" void kernel_launch(void* const* d_in, const int* in_sizes, int n_in,
                              void* d_out, int out_size, void* d_ws,
                              size_t ws_size, hipStream_t stream) {
  (void)in_sizes; (void)n_in; (void)out_size; (void)ws_size;
  const float* x      = (const float*)d_in[0];
  const float* w_attn = (const float*)d_in[1];
  const float* w_proj = (const float*)d_in[2];
  const int*   amask  = (const int*)d_in[3];
  float* out = (float*)d_out;

  // Workspace layout (bytes):
  //   [0,        16 MiB)  xh  (x as f16)           -- later reused as y (f16)
  //   [16 MiB,   40 MiB)  wah (w_attn as f16)
  //   [40 MiB,   48 MiB)  wph (w_proj as f16)
  //   [48 MiB,   96 MiB)  qkv (f16, [3][B][H][T][HD])
  char* ws = (char*)d_ws;
  _Float16* xh  = (_Float16*)(ws);
  _Float16* wah = (_Float16*)(ws + (size_t)16 * 1024 * 1024);
  _Float16* wph = (_Float16*)(ws + (size_t)40 * 1024 * 1024);
  _Float16* qkv = (_Float16*)(ws + (size_t)48 * 1024 * 1024);
  _Float16* y   = xh;  // x (f16) is dead after GEMM1; reuse for attention out

  const int n4x = B_ * T_ * C_ / 4;          // 2097152
  const int n4a = 3 * C_ * C_ / 4;           // 3145728
  const int n4p = C_ * C_ / 4;               // 1048576
  cvt_kernel<<<n4x / 256, 256, 0, stream>>>(x, xh, n4x);
  cvt_kernel<<<n4a / 256, 256, 0, stream>>>(w_attn, wah, n4a);
  cvt_kernel<<<n4p / 256, 256, 0, stream>>>(w_proj, wph, n4p);

  const dim3 blk(128);
  qkv_gemm_kernel<<<dim3(3 * C_ / BN, B_ * T_ / BM), blk, 0, stream>>>(
      xh, wah, qkv);
  attention_kernel<<<dim3(B_ * H_ * (T_ / QB)), blk, 0, stream>>>(
      qkv, amask, y);
  proj_gemm_kernel<<<dim3(C_ / BN, B_ * T_ / BM), blk, 0, stream>>>(
      y, wph, out);
}
